// RNN_4183298147162
// MI455X (gfx1250) — matrix operand order repaired and checked by reading, not compiled, not run on hardware
//
#include <hip/hip_runtime.h>
#include <hip/hip_bf16.h>

typedef __bf16 bf16_t;
typedef __attribute__((ext_vector_type(16))) __bf16 v16bf;
typedef __attribute__((ext_vector_type(8)))  __bf16 v8bf;
typedef __attribute__((ext_vector_type(4)))  __bf16 v4bf;
typedef __attribute__((ext_vector_type(8)))  float  v8f;
typedef int v4i32 __attribute__((vector_size(16)));   // matches builtin param type

#define AS1 __attribute__((address_space(1)))
#define AS3 __attribute__((address_space(3)))

#define HID     768
#define BATCH   256
#define INDIM   129
#define XPAD    160
#define KTOT    928          // 768 (h) + 160 (x, padded from 129)
#define KCHUNKS 29           // 928 / 32
#define KH_CH   24           // 768 / 32 chunks come from h
#define TSTEPS  512
#define OUTD    128

// LDS layout (bytes): double-buffered GEMM stage, then reused for gates.
#define A_BYTES   (32 * 40 * 2)            // 2560
#define B_BYTES   (4 * 64 * 40 * 2)        // 20480
#define BUF_BYTES (A_BYTES + B_BYTES)      // 23040
#define SMEM_BYTES (2 * BUF_BYTES)         // 46080 (> gates 34816)

#if defined(__AMDGCN__) && __has_builtin(__builtin_amdgcn_global_load_async_to_lds_b128)
#define HAS_ASYNC_LDS 1
#else
#define HAS_ASYNC_LDS 0
#endif

#if HAS_ASYNC_LDS
__device__ __forceinline__ void async_cp16(const bf16_t* gsrc, bf16_t* lds_dst) {
    // arg0: generic v4i* (per compiler diagnostic); arg1: LDS pointer —
    // AS(3) representation is the low 32 bits of the generic address.
    __builtin_amdgcn_global_load_async_to_lds_b128(
        (v4i32*)gsrc,
        (AS3 v4i32*)(unsigned int)(unsigned long long)lds_dst,
        0, 0);
}
#endif

template <int N>
__device__ __forceinline__ void wait_async() {
#if defined(__AMDGCN__)
#if __has_builtin(__builtin_amdgcn_s_wait_asynccnt)
    __builtin_amdgcn_s_wait_asynccnt(N);
#else
    asm volatile("s_wait_asynccnt %0" ::"i"(N) : "memory");
#endif
#endif
}

// Pack two 8-element bf16 halves into the 16-element WMMA fragment register.
__device__ __forceinline__ v16bf pack16(const bf16_t* lo, const bf16_t* hi) {
    v8bf a = *(const v8bf*)lo;
    v8bf b = *(const v8bf*)hi;
    return __builtin_shufflevector(a, b, 0,1,2,3,4,5,6,7,8,9,10,11,12,13,14,15);
}

__device__ __forceinline__ float sigmoidf_(float x) {
    return 1.0f / (1.0f + __expf(-x));
}

// ---------------------------------------------------------------- prep kernels
__global__ void prep_wcomb(const float* __restrict__ Whh,
                           const float* __restrict__ Wih,
                           bf16_t* __restrict__ Wcomb) {
    int idx = blockIdx.x * 256 + threadIdx.x;          // 3072*928 elements
    if (idx >= 3072 * KTOT) return;
    int r = idx / KTOT, k = idx % KTOT;
    float v = 0.0f;
    if (k < HID)            v = Whh[r * HID + k];
    else if (k < HID+INDIM) v = Wih[r * INDIM + (k - HID)];
    Wcomb[idx] = (bf16_t)v;
}

__global__ void prep_seq(const float* __restrict__ seq, bf16_t* __restrict__ xseq) {
    int idx = blockIdx.x * 256 + threadIdx.x;          // 512*256*160 elements
    if (idx >= TSTEPS * BATCH * XPAD) return;
    int k = idx % XPAD, tb = idx / XPAD;
    float v = (k < INDIM) ? seq[tb * INDIM + k] : 0.0f;
    xseq[idx] = (bf16_t)v;
}

__global__ void prep_bias(const float* __restrict__ bih, const float* __restrict__ bhh,
                          float* __restrict__ bias) {
    int idx = blockIdx.x * 256 + threadIdx.x;          // 3072
    if (idx < 4 * HID) bias[idx] = bih[idx] + bhh[idx];
}

__global__ void prep_wdec(const float* __restrict__ Wdec, bf16_t* __restrict__ Wdec_bf) {
    int idx = blockIdx.x * 256 + threadIdx.x;          // 128*768
    if (idx < OUTD * HID) Wdec_bf[idx] = (bf16_t)Wdec[idx];
}

__global__ void init_state(bf16_t* __restrict__ h0, bf16_t* __restrict__ h1,
                           float* __restrict__ c) {
    int idx = blockIdx.x * 256 + threadIdx.x;          // 256*768
    if (idx < BATCH * HID) { h0[idx] = (bf16_t)0.0f; h1[idx] = (bf16_t)0.0f; c[idx] = 0.0f; }
}

// ------------------------------------------------------------ fused LSTM step
// grid = (12, 8): blockIdx.x -> 64 hidden units, blockIdx.y -> 32 batch rows.
// 8 waves: wave = gate (w&3) x batch-half (w>>2). Each wave: 4x 16x16 f32 accs.
// gates = [h_prev | x_t] @ Wcomb^T + bias, then i,f,g,o -> c,h update.
// K-chunks (32 wide) are double-buffered in LDS; staging uses the CDNA5 async
// global->LDS path (ASYNCcnt) when available, overlapping copy with WMMA.
__global__ __launch_bounds__(256) void lstm_step_kernel(
    const bf16_t* __restrict__ h_prev, bf16_t* __restrict__ h_next,
    float* __restrict__ c_state, const bf16_t* __restrict__ x_t,
    const bf16_t* __restrict__ Wcomb, const float* __restrict__ bias) {

    __shared__ __align__(16) char smem[SMEM_BYTES];
    float* G_lds = (float*)smem;                        // [4][32][68] f32 (after GEMM)

    const int jb   = blockIdx.x;            // hidden-block (64 units)
    const int bb   = blockIdx.y;            // batch-block (32 rows)
    const int tid  = threadIdx.x;
    const int wave = tid >> 5;
    const int lane = tid & 31;
    const int g    = wave & 3;              // gate i/f/g/o
    const int bh   = wave >> 2;             // batch half (16 rows)

    v8f acc0 = {}, acc1 = {}, acc2 = {}, acc3 = {};

    // fragment lane addressing (documented 16-bit WMMA layouts, wave32)
    const int arow = bh * 16 + (lane & 15);     // A: row m = lane%16
    const int ak   = (lane >> 4) * 8;           // A: k sub-offset per lane half
    const int bn   = lane & 15;                 // B: col n = lane%16
    const int bk   = (lane >> 4) * 8;           // B: k sub-offset per lane half

    // per-iteration staging of one 32-wide K chunk into LDS buffer `buf`
    auto stage = [&](int ch, char* buf) {
        const int kb = ch * 32;
        bf16_t* A_lds = (bf16_t*)buf;
        bf16_t* B_lds = (bf16_t*)(buf + A_BYTES);
#if HAS_ASYNC_LDS
        if (tid < 128) {                     // A: 32 rows x 32 k = 128 x 16B
            int r = tid >> 2, u = (tid & 3) * 8;
            const bf16_t* src = (ch < KH_CH)
                ? (h_prev + (size_t)(bb * 32 + r) * HID  + kb + u)
                : (x_t    + (size_t)(bb * 32 + r) * XPAD + (kb - HID) + u);
            async_cp16(src, A_lds + r * 40 + u);
        }
        {                                    // B: 4 gates x 64 rows x 32 k
            int g2 = tid >> 6, r = tid & 63;
            const bf16_t* src = Wcomb + (size_t)(g2 * HID + jb * 64 + r) * KTOT + kb;
            bf16_t* dst = B_lds + (g2 * 64 + r) * 40;
            #pragma unroll
            for (int u = 0; u < 4; ++u)
                async_cp16(src + u * 8, dst + u * 8);
        }
#else
        {
            int r = tid >> 3, c4 = (tid & 7) * 4;
            const bf16_t* src = (ch < KH_CH)
                ? (h_prev + (size_t)(bb * 32 + r) * HID  + kb + c4)
                : (x_t    + (size_t)(bb * 32 + r) * XPAD + (kb - HID) + c4);
            *(v4bf*)(A_lds + r * 40 + c4) = *(const v4bf*)src;
        }
        {
            int g2 = tid >> 6, r = tid & 63;
            const bf16_t* src = Wcomb + (size_t)(g2 * HID + jb * 64 + r) * KTOT + kb;
            bf16_t* dst = B_lds + (g2 * 64 + r) * 40;
            *(v8bf*)(dst + 0)  = *(const v8bf*)(src + 0);
            *(v8bf*)(dst + 8)  = *(const v8bf*)(src + 8);
            *(v8bf*)(dst + 16) = *(const v8bf*)(src + 16);
            *(v8bf*)(dst + 24) = *(const v8bf*)(src + 24);
        }
#endif
    };

    stage(0, smem);                                      // prologue: chunk 0 -> buf0
    for (int ch = 0; ch < KCHUNKS; ++ch) {
        char* curb = smem + (size_t)(ch & 1) * BUF_BYTES;
        char* nxtb = smem + (size_t)((ch & 1) ^ 1) * BUF_BYTES;
        if (ch + 1 < KCHUNKS) {
            stage(ch + 1, nxtb);                         // overlap copy with compute
#if HAS_ASYNC_LDS
            // only the freshly issued chunk may remain outstanding:
            // waves 0-3 issue 5 async instrs/chunk (A+4B), waves 4-7 issue 4.
            if (wave < 4) wait_async<5>(); else wait_async<4>();
#endif
        } else {
#if HAS_ASYNC_LDS
            wait_async<0>();
#endif
        }
        __syncthreads();                                 // chunk `ch` visible to all

        bf16_t* A_lds = (bf16_t*)curb;
        bf16_t* B_lds = (bf16_t*)(curb + A_BYTES);
        v16bf a = pack16(A_lds + arow * 40 + ak, A_lds + arow * 40 + 16 + ak);
        const bf16_t* bbase = B_lds + g * 64 * 40;
        v16bf b0 = pack16(bbase + (bn     ) * 40 + bk, bbase + (bn     ) * 40 + bk + 16);
        v16bf b1 = pack16(bbase + (16 + bn) * 40 + bk, bbase + (16 + bn) * 40 + bk + 16);
        v16bf b2 = pack16(bbase + (32 + bn) * 40 + bk, bbase + (32 + bn) * 40 + bk + 16);
        v16bf b3 = pack16(bbase + (48 + bn) * 40 + bk, bbase + (48 + bn) * 40 + bk + 16);
        acc0 = __builtin_amdgcn_wmma_f32_16x16x32_bf16(false, a, false, b0, (short)0, acc0, false, false);
        acc1 = __builtin_amdgcn_wmma_f32_16x16x32_bf16(false, a, false, b1, (short)0, acc1, false, false);
        acc2 = __builtin_amdgcn_wmma_f32_16x16x32_bf16(false, a, false, b2, (short)0, acc2, false, false);
        acc3 = __builtin_amdgcn_wmma_f32_16x16x32_bf16(false, a, false, b3, (short)0, acc3, false, false);

        __syncthreads();   // all reads of `curb` done before it is restaged
    }

    // GEMM LDS dead; reuse as gate staging
    {
        const int m0 = (lane >> 4) * 8;     // C/D layout: lanes 16-31 hold rows 8..15
        const int n  = lane & 15;
        v8f accs[4] = {acc0, acc1, acc2, acc3};
        #pragma unroll
        for (int jt = 0; jt < 4; ++jt) {
            float bv = bias[g * HID + jb * 64 + jt * 16 + n];
            #pragma unroll
            for (int r = 0; r < 8; ++r) {
                G_lds[(g * 32 + bh * 16 + m0 + r) * 68 + jt * 16 + n] = accs[jt][r] + bv;
            }
        }
    }
    __syncthreads();

    // cell update: 32x64 elements, 8 per thread (coalesced over j)
    #pragma unroll
    for (int e = 0; e < 8; ++e) {
        int idx = e * 256 + tid;            // 0..2047
        int r = idx >> 6, j = idx & 63;
        float iv = sigmoidf_(G_lds[(0 * 32 + r) * 68 + j]);
        float fv = sigmoidf_(G_lds[(1 * 32 + r) * 68 + j]);
        float gv = tanhf    (G_lds[(2 * 32 + r) * 68 + j]);
        float ov = sigmoidf_(G_lds[(3 * 32 + r) * 68 + j]);
        size_t gi = (size_t)(bb * 32 + r) * HID + jb * 64 + j;
        float cn = fv * c_state[gi] + iv * gv;
        c_state[gi] = cn;
        h_next[gi]  = (bf16_t)(ov * tanhf(cn));
    }
}

// ------------------------------------------------------------- decoder GEMM
// grid = (16, 8), 1 wave per block: 16x16 output tile of  h @ W_dec^T + b_dec.
__global__ __launch_bounds__(32) void decode_kernel(
    const bf16_t* __restrict__ hbuf, const bf16_t* __restrict__ Wdec,
    const float* __restrict__ b_dec, float* __restrict__ out) {
    const int lane  = threadIdx.x & 31;
    const int bbase = blockIdx.x * 16;
    const int obase = blockIdx.y * 16;
    const int arow  = bbase + (lane & 15);
    const int ak    = (lane >> 4) * 8;
    const int brow  = obase + (lane & 15);
    const int bk    = (lane >> 4) * 8;

    v8f acc = {};
    #pragma unroll 4
    for (int kb = 0; kb < HID; kb += 32) {
        v16bf a = pack16(hbuf + (size_t)arow * HID + kb + ak,
                         hbuf + (size_t)arow * HID + kb + 16 + ak);
        v16bf b = pack16(Wdec + (size_t)brow * HID + kb + bk,
                         Wdec + (size_t)brow * HID + kb + bk + 16);
        acc = __builtin_amdgcn_wmma_f32_16x16x32_bf16(false, a, false, b, (short)0, acc, false, false);
    }
    const int n  = lane & 15;
    const int m0 = (lane >> 4) * 8;
    float bv = b_dec[obase + n];
    #pragma unroll
    for (int r = 0; r < 8; ++r)
        out[(size_t)(bbase + m0 + r) * OUTD + obase + n] = acc[r] + bv;
}

// ---------------------------------------------------------------- host side
extern "C" void kernel_launch(void* const* d_in, const int* in_sizes, int n_in,
                              void* d_out, int out_size, void* d_ws, size_t ws_size,
                              hipStream_t stream) {
    const float* seq  = (const float*)d_in[0];
    const float* Wih  = (const float*)d_in[1];
    const float* Whh  = (const float*)d_in[2];
    const float* bih  = (const float*)d_in[3];
    const float* bhh  = (const float*)d_in[4];
    const float* Wdec = (const float*)d_in[5];
    const float* bdec = (const float*)d_in[6];
    float* out = (float*)d_out;

    char* ws = (char*)d_ws;
    size_t off = 0;
    auto carve = [&](size_t bytes) -> void* {
        void* p = ws + off;
        off = (off + bytes + 255) & ~(size_t)255;
        return p;
    };
    bf16_t* Wcomb   = (bf16_t*)carve((size_t)3072 * KTOT * 2);          // 5.7 MB
    bf16_t* Wdec_bf = (bf16_t*)carve((size_t)OUTD * HID * 2);           // 196 KB
    float*  bias    = (float*) carve((size_t)4 * HID * 4);              // 12 KB
    bf16_t* xseq    = (bf16_t*)carve((size_t)TSTEPS * BATCH * XPAD * 2);// 42 MB
    bf16_t* h0      = (bf16_t*)carve((size_t)BATCH * HID * 2);
    bf16_t* h1      = (bf16_t*)carve((size_t)BATCH * HID * 2);
    float*  c       = (float*) carve((size_t)BATCH * HID * 4);

    prep_wcomb<<<(3072 * KTOT + 255) / 256, 256, 0, stream>>>(Whh, Wih, Wcomb);
    prep_seq  <<<(TSTEPS * BATCH * XPAD + 255) / 256, 256, 0, stream>>>(seq, xseq);
    prep_bias <<<(4 * HID + 255) / 256, 256, 0, stream>>>(bih, bhh, bias);
    prep_wdec <<<(OUTD * HID + 255) / 256, 256, 0, stream>>>(Wdec, Wdec_bf);
    init_state<<<(BATCH * HID + 255) / 256, 256, 0, stream>>>(h0, h1, c);

    bf16_t* hb[2] = {h0, h1};
    for (int t = 0; t < TSTEPS; ++t) {
        lstm_step_kernel<<<dim3(12, 8), 256, 0, stream>>>(
            hb[t & 1], hb[(t + 1) & 1], c,
            xseq + (size_t)t * BATCH * XPAD, Wcomb, bias);
        if (t >= TSTEPS / 2) {
            decode_kernel<<<dim3(16, 8), 32, 0, stream>>>(
                hb[(t + 1) & 1], Wdec_bf, bdec,
                out + (size_t)(t - TSTEPS / 2) * BATCH * OUTD);
        }
    }
}
